// KagomeSignNetwork_29755533427567
// MI455X (gfx1250) — matrix operand order repaired and checked by reading, not compiled, not run on hardware
//
#include <hip/hip_runtime.h>
#include <hip/hip_bf16.h>

typedef __attribute__((ext_vector_type(16))) _Float16 v16h;
typedef __attribute__((ext_vector_type(8)))  _Float16 v8h;
typedef __attribute__((ext_vector_type(8)))  float    v8f;

#define NSITE 36
#define NK    15
#define CCH   32
#define TS    16   // samples per workgroup (= WMMA M)

// Graph topology is a compile-time constant in the reference (ADJ/SUBL/RED
// literals). Bake it so neighbor indices become uniform scalar (s_load) reads.
__constant__ int nbrC[NSITE*NK] = {
28,29,31,32,33,34,35,0,16,3,4,5,20,7,8,
14,15,30,19,35,25,26,1,2,3,29,6,34,9,10,
15,30,31,19,35,26,1,2,3,4,6,7,9,10,11,
30,31,32,35,0,1,2,3,4,5,6,7,10,11,12,
31,32,33,35,0,2,3,4,5,20,7,8,11,12,13,
32,33,34,0,16,3,4,5,20,21,7,8,12,13,14,
19,35,26,1,2,3,29,6,7,34,9,10,11,16,17,
35,0,2,3,4,5,6,7,8,10,11,12,13,17,18,
0,16,4,5,20,21,7,8,27,12,13,14,15,18,19,
26,1,2,29,6,33,34,9,10,11,16,17,20,21,22,
1,2,3,6,7,34,9,10,11,12,16,17,21,22,23,
2,3,4,6,7,9,10,11,12,13,17,18,22,23,24,
3,4,5,7,8,10,11,12,13,14,17,18,23,24,25,
4,5,20,7,8,11,12,13,14,15,18,19,24,25,26,
5,20,21,8,27,12,13,14,15,30,18,19,25,26,1,
20,21,22,8,27,13,14,15,30,31,19,35,26,1,2,
29,6,33,34,9,10,0,16,17,5,20,21,22,8,27,
6,7,9,10,11,12,16,17,18,21,22,23,24,27,28,
7,8,11,12,13,14,17,18,19,23,24,25,26,28,29,
8,27,13,14,15,30,18,19,35,25,26,1,2,29,6,
33,34,9,0,16,4,5,20,21,22,8,27,13,14,15,
34,9,10,16,17,5,20,21,22,23,8,27,14,15,30,
9,10,11,16,17,20,21,22,23,24,27,28,15,30,31,
10,11,12,17,18,21,22,23,24,25,27,28,30,31,32,
11,12,13,17,18,22,23,24,25,26,28,29,31,32,33,
12,13,14,18,19,23,24,25,26,1,28,29,32,33,34,
13,14,15,18,19,24,25,26,1,2,29,6,33,34,9,
16,17,20,21,22,23,8,27,28,14,15,30,31,19,35,
17,18,22,23,24,25,27,28,29,30,31,32,33,35,0,
18,19,24,25,26,1,28,29,6,32,33,34,9,0,16,
21,22,23,27,28,14,15,30,31,32,19,35,1,2,3,
22,23,24,27,28,15,30,31,32,33,35,0,2,3,4,
23,24,25,28,29,30,31,32,33,34,35,0,3,4,5,
24,25,26,28,29,31,32,33,34,9,0,16,4,5,20,
25,26,1,29,6,32,33,34,9,10,0,16,5,20,21,
27,28,15,30,31,32,19,35,0,1,2,3,4,6,7};

__constant__ int sublC[NSITE] = {
0,2,1,2,1,2,0,0,0,1,2,1,2,1,2,1,0,0,0,0,1,2,1,2,1,2,1,0,0,0,2,1,2,1,2,0};

__constant__ int redC[36] = {   // RED[3][12]: sites per sublattice type
0,6,7,8,16,17,18,19,27,28,29,35,
2,4,9,11,13,15,20,22,24,26,31,33,
1,3,5,10,12,14,21,23,25,30,32,34};

// Dynamic LDS layout (bytes)
constexpr int OFF_HA   = 0;
constexpr int OFF_HB   = OFF_HA + NSITE*TS*CCH*2;          // 36864
constexpr int OFF_WB   = OFF_HB + NSITE*TS*CCH*2;          // 73728
constexpr int OFF_BIAS = OFF_WB + 3*NK*CCH*CCH*2;          // 165888
constexpr int OFF_X    = OFF_BIAS + 96*4;                  // 166272 (8B aligned)
constexpr int SMEM_BYTES = OFF_X + TS*8;                   // 166400

extern "C" __global__ __launch_bounds__(256)
void kagome_fused(const long long* __restrict__ x,
                  const float* __restrict__ W0, const float* __restrict__ b0,
                  const float* __restrict__ W1, const float* __restrict__ b1,
                  const float* __restrict__ W2, const float* __restrict__ b2,
                  const float* __restrict__ W3, const float* __restrict__ b3,
                  const float* __restrict__ Wt, const float* __restrict__ bt,
                  float* __restrict__ out)
{
    extern __shared__ char smem[];
    _Float16* hA    = (_Float16*)(smem + OFF_HA);
    _Float16* hB    = (_Float16*)(smem + OFF_HB);
    _Float16* WB    = (_Float16*)(smem + OFF_WB);   // [t][k][o][c] f16 (layer0: [t][o][k32])
    float*    biasL = (float*)   (smem + OFF_BIAS);
    long long* xL   = (long long*)(smem + OFF_X);

    const int tid  = threadIdx.x;
    const int wg   = blockIdx.x;
    const int wave = tid >> 5;
    const int lane = tid & 31;
    const int sel  = lane >> 4;     // which K-halves this lane holds
    const int lan16 = lane & 15;    // sample row (A) / output column (B,C)

    // ---- stage spins, layer-0 weights/bias ----
    if (tid < TS) xL[tid] = x[(long long)wg*TS + tid];
    // layer-0 B: WB0[t][o][k], k padded 15->32 with zeros. W0 shape (3,32,1,15)
    for (int i = tid; i < 3*CCH*CCH; i += 256) {
        int k = i & 31; int o = (i >> 5) & 31; int t = i >> 10;
        WB[i] = (k < NK) ? (_Float16)W0[(t*CCH + o)*NK + k] : (_Float16)0.0f;
    }
    for (int i = tid; i < 96; i += 256) biasL[i] = b0[i];
    __syncthreads();

    // ---- layer 0: spins (K=15 padded) -> hA ----
    {
        const long long xv = xL[lan16];
        for (int i = 0; i < 9; ++i) {
            int tile = wave*9 + i;
            int n  = __builtin_amdgcn_readfirstlane(tile >> 1);
            int half = tile & 1;
            int t  = __builtin_amdgcn_readfirstlane(sublC[n]);
            int o  = half*16 + lan16;
            // A order used here (contiguous K, matches B): lanes<16 K=0..15, lanes>=16 K=16..31
            v16h a;
            #pragma unroll
            for (int e = 0; e < 16; ++e) {
                int K = e + sel*16;
                int site = nbrC[n*NK + (K < NK ? K : 0)];   // clamped: branchless
                _Float16 sv = ((xv >> site) & 1) ? (_Float16)1.0f : (_Float16)(-1.0f);
                a[e] = (K < NK) ? sv : (_Float16)0.0f;
            }
            // B lane layout: lanes<16 hold K=0..15 of column N, lanes>=16 K=16..31
            v16h bm = *(const v16h*)(&WB[(t*CCH + o)*CCH + sel*16]);
            v8f acc = {};
            acc = __builtin_amdgcn_wmma_f32_16x16x32_f16(false, a, false, bm,
                                                         (short)0, acc, false, false);
            float bv = biasL[t*CCH + o];
            #pragma unroll
            for (int r = 0; r < 8; ++r) {
                int m = r + sel*8;                 // C layout: VGPR r, lanes>=16 -> M=r+8
                float v = acc[r] + bv;
                hA[(n*TS + m)*CCH + o] = (_Float16)(v > 0.f ? v : 0.f);
            }
        }
    }
    __syncthreads();

    // ---- layers 1..3: (B x 480) @ (480 x 32) per site via 15 WMMA K-steps ----
    const float* Wl[3] = {W1, W2, W3};
    const float* bl[3] = {b1, b2, b3};
    _Float16* hin = hA; _Float16* hout = hB;
    const int laneAoff = lan16*CCH + sel*16;       // lane-constant A offset (halves)
    for (int layer = 0; layer < 3; ++layer) {
        // stage WB[t][k][o][c] <- W[t][o][c][k]  (3,32,32,15) f32 -> f16
        const float* Wg = Wl[layer];
        for (int idx = tid; idx < 3*NK*CCH*CCH; idx += 256) {
            int c = idx & 31;
            int o = (idx >> 5) & 31;
            int r = idx >> 10;
            int k = r % NK;
            int t = r / NK;
            WB[idx] = (_Float16)Wg[((t*CCH + o)*CCH + c)*NK + k];
        }
        for (int i = tid; i < 96; i += 256) biasL[i] = bl[layer][i];
        __syncthreads();

        for (int i = 0; i < 9; ++i) {
            int tile = wave*9 + i;
            int n  = __builtin_amdgcn_readfirstlane(tile >> 1);
            int half = tile & 1;
            int t  = __builtin_amdgcn_readfirstlane(sublC[n]);
            int o  = half*16 + lan16;

            // hoist neighbor indices into SGPRs (uniform address -> s_load)
            int js[NK];
            #pragma unroll
            for (int k = 0; k < NK; ++k) js[k] = nbrC[n*NK + k];

            const _Float16* bbase = &WB[(t*NK*CCH + o)*CCH + sel*16]; // +k*1024 halves
            v8f acc = {};
            #pragma unroll
            for (int k = 0; k < NK; ++k) {
                // A: h[j][sample=lan16][c], lane<16 c=0..15, lane>=16 c=16..31
                const _Float16* ab = hin + js[k]*(TS*CCH) + laneAoff;
                v8h lo = *(const v8h*)(ab);
                v8h hi = *(const v8h*)(ab + 8);
                v16h a = __builtin_shufflevector(lo, hi,
                         0,1,2,3,4,5,6,7,8,9,10,11,12,13,14,15);
                // B: WB[t][k][o][c], 32B contiguous per lane; k steps = DS imm offsets
                v16h bm = *(const v16h*)(bbase + k*(CCH*CCH));
                acc = __builtin_amdgcn_wmma_f32_16x16x32_f16(false, a, false, bm,
                                                             (short)0, acc, false, false);
            }
            float bv = biasL[t*CCH + o];
            #pragma unroll
            for (int r = 0; r < 8; ++r) {
                int m = r + sel*8;
                float v = acc[r] + bv;
                hout[(n*TS + m)*CCH + o] = (_Float16)(v > 0.f ? v : 0.f);
            }
        }
        __syncthreads();
        _Float16* tmp = hin; hin = hout; hout = tmp;
    }
    // after 3 swaps hin == hB holds final features

    // ---- head: sublattice mean-pool (B,32,3) -> flat c*3+t -> @ Wt.T + bt ----
    if (tid < 32) {
        int b  = tid & 15;
        int jo = tid >> 4;
        float acc = bt[jo];
        for (int t = 0; t < 3; ++t) {
            for (int c = 0; c < CCH; ++c) {
                float s = 0.f;
                #pragma unroll
                for (int e = 0; e < 12; ++e) {
                    int site = redC[t*12 + e];
                    s += (float)hin[(site*TS + b)*CCH + c];
                }
                acc += (s * (1.0f/12.0f)) * Wt[jo*96 + (c*3 + t)];
            }
        }
        out[((long long)wg*TS + b)*2 + jo] = acc;
    }
}

extern "C" void kernel_launch(void* const* d_in, const int* in_sizes, int n_in,
                              void* d_out, int out_size, void* d_ws, size_t ws_size,
                              hipStream_t stream) {
    const long long* x  = (const long long*)d_in[0];
    const float* W0 = (const float*)d_in[1];
    const float* b0 = (const float*)d_in[2];
    const float* W1 = (const float*)d_in[3];
    const float* b1 = (const float*)d_in[4];
    const float* W2 = (const float*)d_in[5];
    const float* b2 = (const float*)d_in[6];
    const float* W3 = (const float*)d_in[7];
    const float* b3 = (const float*)d_in[8];
    const float* Wt = (const float*)d_in[9];
    const float* bt = (const float*)d_in[10];
    float* out = (float*)d_out;

    (void)d_ws; (void)ws_size; (void)n_in; (void)out_size;

    int B = in_sizes[0];
    int nwg = (B + TS - 1) / TS;

    // allow >64KB dynamic LDS (CDNA5 WGP supports up to 320KB per workgroup)
    static_assert(SMEM_BYTES < 320*1024, "LDS budget");
    hipFuncSetAttribute((const void*)kagome_fused,
                        hipFuncAttributeMaxDynamicSharedMemorySize, SMEM_BYTES);

    kagome_fused<<<nwg, 256, SMEM_BYTES, stream>>>(
        x, W0, b0, W1, b1, W2, b2, W3, b3, Wt, bt, out);
}
